// DCN_89859305767621
// MI455X (gfx1250) — compile-verified, matrix operand order checked by reading, not run on hardware
//
#include <hip/hip_runtime.h>
#include <hip/hip_bf16.h>
#include <math.h>

typedef __attribute__((ext_vector_type(16))) _Float16 v16h;
typedef __attribute__((ext_vector_type(8)))  _Float16 v8h;
typedef __attribute__((ext_vector_type(8)))  float    v8f;

#define NDISC 38
#define NCONT 90
#define NCSL  30
#define CCH   4
#define NH    128
#define NH0   44
#define KP0   64
#define NH1   172
#define NH1P  176
#define KP1   192
#define NH2   344
#define KP2   352

__device__ __forceinline__ float leaky01(float v) { return v > 0.f ? v : 0.01f * v; }

// ---------------------------------------------------------------------------
// Weight prep: fp32 W[K,Nout] -> f16 Wt[NoutPad, Kpad] (transposed, zero-pad)
// ---------------------------------------------------------------------------
__global__ void k_wprep(const float* __restrict__ W, _Float16* __restrict__ Wt,
                        int K, int Nout, int Kpad, int NoutPad) {
    int idx = blockIdx.x * blockDim.x + threadIdx.x;
    int tot = Kpad * NoutPad;
    if (idx >= tot) return;
    int j = idx / Kpad, k = idx - j * Kpad;
    float v = (j < Nout && k < K) ? W[(size_t)k * Nout + j] : 0.f;
    Wt[idx] = (_Float16)v;
}

// ---------------------------------------------------------------------------
// Feature embedding: argmax(one-hot dow) + raw discrete + per-slice cont MLP
// Writes: x44 f16 [N,KP0] (g0 GEMM input, zero-padded),
//         fused32 [N,NH1] cols 0..43, fused16 [N,KP1] cols 0..43
// ---------------------------------------------------------------------------
__global__ void k_features(const int* __restrict__ dx, const float* __restrict__ cx,
                           const float* __restrict__ Wc, const float* __restrict__ bc,
                           _Float16* __restrict__ x44, float* __restrict__ fused32,
                           _Float16* __restrict__ fused16, int N) {
    int n = blockIdx.x * blockDim.x + threadIdx.x;
    if (n >= N) return;
    const int* dr = dx + (size_t)n * NDISC;
    int best = dr[0], bi = 0;
    for (int i = 1; i < 7; i++) { int v = dr[i]; if (v > best) { best = v; bi = i; } }
    float feat[NH0];
    feat[0] = (float)bi;
    for (int i = 0; i < 31; i++) feat[1 + i] = (float)dr[7 + i];
    const float* cr = cx + (size_t)n * NCONT;
    for (int s = 0; s < 3; s++)
        for (int c = 0; c < CCH; c++) {
            float acc = bc[c];
            for (int k = 0; k < NCSL; k++) acc += cr[s * NCSL + k] * Wc[k * CCH + c];
            feat[32 + s * CCH + c] = acc;
        }
    _Float16* xr  = x44 + (size_t)n * KP0;
    float*    f3r = fused32 + (size_t)n * NH1;
    _Float16* f1r = fused16 + (size_t)n * KP1;
    for (int i = 0; i < NH0; i++) {
        xr[i] = (_Float16)feat[i];
        f3r[i] = feat[i];
        f1r[i] = (_Float16)feat[i];
    }
    for (int i = NH0; i < KP0; i++) xr[i] = (_Float16)0.f;
}

// ---------------------------------------------------------------------------
// Degree: atomic count of dst, then deg+1 (self loop) and rsqrt
// ---------------------------------------------------------------------------
__global__ void k_deg(const int* __restrict__ ei, float* __restrict__ deg, int E) {
    int e = blockIdx.x * blockDim.x + threadIdx.x;
    if (e >= E) return;
    atomicAdd(&deg[ei[(size_t)E + e]], 1.0f);
}
__global__ void k_degfin(float* __restrict__ deg, float* __restrict__ dis, int N) {
    int n = blockIdx.x * blockDim.x + threadIdx.x;
    if (n >= N) return;
    float d = deg[n] + 1.0f;
    deg[n] = d;
    dis[n] = rsqrtf(d);
}

// ---------------------------------------------------------------------------
// WMMA GEMM: C[M,Nout] = act(A[M,Kpad] @ Wt^T + bias)
// A row-major f16 (stride lda), Wt row-major f16 [NoutPad, kpad].
// One wave computes one 16x16 tile; K stepped by 32 via v_wmma_f32_16x16x32_f16.
// Per lane per K-step: 2x b128 loads of A, 2x b128 loads of Wt.
// ---------------------------------------------------------------------------
__global__ void k_gemm(const _Float16* __restrict__ A, int lda,
                       const _Float16* __restrict__ Wt, int kpad,
                       const float* __restrict__ bias, int nlogic,
                       _Float16* __restrict__ out16, int ld16, int off16,
                       float* __restrict__ out32, int ld32, int off32,
                       int tiles_n, int total_tiles, int act) {
    int wave = threadIdx.x >> 5;
    int t = blockIdx.x * (blockDim.x >> 5) + wave;
    if (t >= total_tiles) return;                 // wave-uniform: EXEC stays all-1s
    int tm = t / tiles_n, tn = t - tm * tiles_n;
    int lane = threadIdx.x & 31;
    int r = lane & 15, half = lane >> 4;
    const _Float16* arow = A  + (size_t)(tm * 16 + r) * lda;
    const _Float16* brow = Wt + (size_t)(tn * 16 + r) * kpad;
    v8f acc = {};
    for (int kk = 0; kk < kpad; kk += 32) {
        v8h alo = *reinterpret_cast<const v8h*>(arow + kk + 8 * half);
        v8h ahi = *reinterpret_cast<const v8h*>(arow + kk + 16 + 8 * half);
        v8h blo = *reinterpret_cast<const v8h*>(brow + kk + 8 * half);
        v8h bhi = *reinterpret_cast<const v8h*>(brow + kk + 16 + 8 * half);
        v16h av, bv;
#pragma unroll
        for (int i = 0; i < 8; i++) { av[i] = alo[i]; av[8 + i] = ahi[i];
                                      bv[i] = blo[i]; bv[8 + i] = bhi[i]; }
        acc = __builtin_amdgcn_wmma_f32_16x16x32_f16(
            false, av, false, bv, (short)0, acc, false, false);
    }
    int col = tn * 16 + r;
    float b = (bias != nullptr && col < nlogic) ? bias[col] : 0.f;
#pragma unroll
    for (int i = 0; i < 8; i++) {
        int row = tm * 16 + i + 8 * half;
        float v = acc[i] + b;
        if (act) v = leaky01(v);
        if (out16) out16[(size_t)row * ld16 + off16 + col] = (_Float16)v;
        if (out32) out32[(size_t)row * ld32 + off32 + col] = v;
    }
}

// ---------------------------------------------------------------------------
// GCN message scatter: wave per edge, lane handles 4 of 128 features.
// agg[dst] += xw[src] * dis[src]*dis[dst]
// ---------------------------------------------------------------------------
__global__ void k_edge(const int* __restrict__ ei, const float* __restrict__ dis,
                       const float* __restrict__ xw, float* __restrict__ agg, int E) {
    int wave = threadIdx.x >> 5;
    int e = blockIdx.x * (blockDim.x >> 5) + wave;
    if (e >= E) return;
    int lane = threadIdx.x & 31;
    int src = ei[e], dst = ei[(size_t)E + e];
    float c = dis[src] * dis[dst];
    const float4 v = *reinterpret_cast<const float4*>(xw + (size_t)src * NH + lane * 4);
    float* ap = agg + (size_t)dst * NH + lane * 4;
    atomicAdd(ap + 0, v.x * c);
    atomicAdd(ap + 1, v.y * c);
    atomicAdd(ap + 2, v.z * c);
    atomicAdd(ap + 3, v.w * c);
}

// ---------------------------------------------------------------------------
// GCN epilogue: out = leaky(agg + xw/deg + bias); optional fused write
// ---------------------------------------------------------------------------
__global__ void k_gcn_epi(const float* __restrict__ agg, const float* __restrict__ xw,
                          const float* __restrict__ deg, const float* __restrict__ bias,
                          _Float16* __restrict__ out16, float* __restrict__ fused32,
                          _Float16* __restrict__ fused16, int N) {
    int t = blockIdx.x * blockDim.x + threadIdx.x;
    if (t >= N * NH) return;
    int n = t / NH, f = t - n * NH;
    float v = leaky01(agg[t] + xw[t] / deg[n] + bias[f]);
    if (out16) out16[t] = (_Float16)v;
    if (fused32) {
        fused32[(size_t)n * NH1 + NH0 + f] = v;
        fused16[(size_t)n * KP1 + NH0 + f] = (_Float16)v;
    }
}

// ---------------------------------------------------------------------------
// Both cross layers fused, wave per row (xl lives in registers, 6 per lane).
// xl = x0*(xl . w_l) + b_l + xl; final xl -> h f16 cols [172, 344)
// ---------------------------------------------------------------------------
__global__ void k_cross(const float* __restrict__ fused32, const float* __restrict__ cw,
                        const float* __restrict__ cb, _Float16* __restrict__ h, int N) {
    int wave = threadIdx.x >> 5;
    int row = blockIdx.x * (blockDim.x >> 5) + wave;
    if (row >= N) return;
    int lane = threadIdx.x & 31;
    float x0v[6], xlv[6];
#pragma unroll
    for (int j = 0; j < 6; j++) {
        int c = lane + 32 * j;
        float v = (c < NH1) ? fused32[(size_t)row * NH1 + c] : 0.f;
        x0v[j] = v; xlv[j] = v;
    }
    for (int l = 0; l < 2; l++) {
        float p = 0.f;
#pragma unroll
        for (int j = 0; j < 6; j++) {
            int c = lane + 32 * j;
            if (c < NH1) p += xlv[j] * cw[l * NH1 + c];
        }
#pragma unroll
        for (int o = 16; o > 0; o >>= 1) p += __shfl_xor(p, o, 32);
#pragma unroll
        for (int j = 0; j < 6; j++) {
            int c = lane + 32 * j;
            if (c < NH1) xlv[j] = x0v[j] * p + cb[l * NH1 + c] + xlv[j];
        }
    }
#pragma unroll
    for (int j = 0; j < 6; j++) {
        int c = lane + 32 * j;
        if (c < NH1) h[(size_t)row * KP2 + NH1 + c] = (_Float16)xlv[j];
    }
}

// ---------------------------------------------------------------------------
// Final matvec + sigmoid, wave per row
// ---------------------------------------------------------------------------
__global__ void k_pred(const _Float16* __restrict__ p1, const float* __restrict__ W2,
                       const float* __restrict__ b2, float* __restrict__ out, int N) {
    int wave = threadIdx.x >> 5;
    int row = blockIdx.x * (blockDim.x >> 5) + wave;
    if (row >= N) return;
    int lane = threadIdx.x & 31;
    float p = 0.f;
    for (int c = lane; c < NH1; c += 32) p += (float)p1[(size_t)row * KP1 + c] * W2[c];
#pragma unroll
    for (int o = 16; o > 0; o >>= 1) p += __shfl_xor(p, o, 32);
    if (lane == 0) out[row] = 1.f / (1.f + expf(-(p + b2[0])));
}

// ---------------------------------------------------------------------------
extern "C" void kernel_launch(void* const* d_in, const int* in_sizes, int n_in,
                              void* d_out, int out_size, void* d_ws, size_t ws_size,
                              hipStream_t stream) {
    const int N = in_sizes[0] / NDISC;     // 100000 (multiple of 16)
    const int E = in_sizes[2] / 2;         // 1600000

    const int*   discrete_x = (const int*)  d_in[0];
    const float* cont_x     = (const float*)d_in[1];
    const int*   edge_index = (const int*)  d_in[2];
    const float* emb_c_W    = (const float*)d_in[5];
    const float* emb_c_b    = (const float*)d_in[6];
    const float* g0_W       = (const float*)d_in[7];
    const float* g0_b       = (const float*)d_in[8];
    const float* gcn1_W     = (const float*)d_in[9];
    const float* gcn1_b     = (const float*)d_in[10];
    const float* gcn2_W     = (const float*)d_in[11];
    const float* gcn2_b     = (const float*)d_in[12];
    const float* dnn_W1     = (const float*)d_in[13];
    const float* dnn_b1     = (const float*)d_in[14];
    const float* dnn_W2     = (const float*)d_in[15];
    const float* dnn_b2     = (const float*)d_in[16];
    const float* cross_w    = (const float*)d_in[17];
    const float* cross_b    = (const float*)d_in[18];
    const float* pred_W1    = (const float*)d_in[19];
    const float* pred_b1    = (const float*)d_in[20];
    const float* pred_W2    = (const float*)d_in[21];
    const float* pred_b2    = (const float*)d_in[22];
    float* out = (float*)d_out;

    // ---- workspace carve-up (256B aligned) ----
    char* base = (char*)d_ws;
    size_t off = 0;
    auto carve = [&](size_t bytes) -> char* {
        char* p = base + off;
        off = (off + bytes + 255) & ~(size_t)255;
        return p;
    };
    _Float16* g0Wt    = (_Float16*)carve((size_t)NH   * KP0 * 2);
    _Float16* gcn1Wt  = (_Float16*)carve((size_t)NH   * NH  * 2);
    _Float16* gcn2Wt  = (_Float16*)carve((size_t)NH   * NH  * 2);
    _Float16* dnnW1t  = (_Float16*)carve((size_t)NH1P * KP1 * 2);
    _Float16* dnnW2t  = (_Float16*)carve((size_t)NH1P * KP1 * 2);
    _Float16* predW1t = (_Float16*)carve((size_t)NH1P * KP2 * 2);
    _Float16* x44     = (_Float16*)carve((size_t)N * KP0 * 2);
    float*    fused32 = (float*)   carve((size_t)N * NH1 * 4);
    _Float16* fused16 = (_Float16*)carve((size_t)N * KP1 * 2);
    float*    deg     = (float*)   carve((size_t)N * 4);
    float*    dis     = (float*)   carve((size_t)N * 4);
    _Float16* xgA     = (_Float16*)carve((size_t)N * NH * 2);
    _Float16* xgB     = (_Float16*)carve((size_t)N * NH * 2);
    char*     region1 = carve((size_t)N * NH * 4);   // xw f32, later d1/p1 f16
    float*    agg     = (float*)   carve((size_t)N * NH * 4);
    _Float16* h       = (_Float16*)carve((size_t)N * KP2 * 2);
    float*    xw = (float*)region1;
    _Float16* d1 = (_Float16*)region1;
    _Float16* p1 = (_Float16*)region1;

    const int tiles_m = N / 16;
    auto gemm = [&](const _Float16* A, int lda, const _Float16* Wt, int kpad,
                    const float* bias, int nlogic,
                    _Float16* o16, int ld16, int off16,
                    float* o32, int ld32, int off32, int tiles_n, int act) {
        int total = tiles_m * tiles_n;
        k_gemm<<<(total + 7) / 8, 256, 0, stream>>>(A, lda, Wt, kpad, bias, nlogic,
                                                    o16, ld16, off16, o32, ld32, off32,
                                                    tiles_n, total, act);
    };

    // ---- weight prep ----
    auto wprep = [&](const float* W, _Float16* Wt, int K, int Nout, int Kpad, int NoutPad) {
        int tot = Kpad * NoutPad;
        k_wprep<<<(tot + 255) / 256, 256, 0, stream>>>(W, Wt, K, Nout, Kpad, NoutPad);
    };
    wprep(g0_W,    g0Wt,    NH0, NH,  KP0, NH);
    wprep(gcn1_W,  gcn1Wt,  NH,  NH,  NH,  NH);
    wprep(gcn2_W,  gcn2Wt,  NH,  NH,  NH,  NH);
    wprep(dnn_W1,  dnnW1t,  NH1, NH1, KP1, NH1P);
    wprep(dnn_W2,  dnnW2t,  NH1, NH1, KP1, NH1P);
    wprep(pred_W1, predW1t, NH2, NH1, KP2, NH1P);

    // ---- zero buffers whose pad columns feed later GEMMs ----
    hipMemsetAsync(deg,     0, (size_t)N * 4,        stream);
    hipMemsetAsync(fused16, 0, (size_t)N * KP1 * 2,  stream);
    hipMemsetAsync(h,       0, (size_t)N * KP2 * 2,  stream);

    // ---- features + degrees ----
    k_features<<<(N + 255) / 256, 256, 0, stream>>>(discrete_x, cont_x, emb_c_W, emb_c_b,
                                                    x44, fused32, fused16, N);
    k_deg<<<(E + 255) / 256, 256, 0, stream>>>(edge_index, deg, E);
    k_degfin<<<(N + 255) / 256, 256, 0, stream>>>(deg, dis, N);

    // ---- g0: x_g = leaky(x44 @ g0_W + b) ----
    gemm(x44, KP0, g0Wt, KP0, g0_b, NH, xgA, NH, 0, nullptr, 0, 0, NH / 16, 1);

    // ---- GCN layer 1 ----
    gemm(xgA, NH, gcn1Wt, NH, nullptr, 0, nullptr, 0, 0, xw, NH, 0, NH / 16, 0);
    hipMemsetAsync(agg, 0, (size_t)N * NH * 4, stream);
    k_edge<<<(E + 7) / 8, 256, 0, stream>>>(edge_index, dis, xw, agg, E);
    k_gcn_epi<<<(N * NH + 255) / 256, 256, 0, stream>>>(agg, xw, deg, gcn1_b,
                                                        xgB, nullptr, nullptr, N);
    // ---- GCN layer 2 (writes fused cols 44..171) ----
    gemm(xgB, NH, gcn2Wt, NH, nullptr, 0, nullptr, 0, 0, xw, NH, 0, NH / 16, 0);
    hipMemsetAsync(agg, 0, (size_t)N * NH * 4, stream);
    k_edge<<<(E + 7) / 8, 256, 0, stream>>>(edge_index, dis, xw, agg, E);
    k_gcn_epi<<<(N * NH + 255) / 256, 256, 0, stream>>>(agg, xw, deg, gcn2_b,
                                                        nullptr, fused32, fused16, N);

    // ---- DNN branch (region1 now reused as f16 d1; zero pad cols first) ----
    hipMemsetAsync(region1, 0, (size_t)N * KP1 * 2, stream);
    gemm(fused16, KP1, dnnW1t, KP1, dnn_b1, NH1, d1, KP1, 0, nullptr, 0, 0, NH1P / 16, 1);
    gemm(d1, KP1, dnnW2t, KP1, dnn_b2, NH1, h, KP2, 0, nullptr, 0, 0, NH1P / 16, 1);

    // ---- Cross branch (both layers fused; writes h cols 172..343) ----
    k_cross<<<(N + 7) / 8, 256, 0, stream>>>(fused32, cross_w, cross_b, h, N);

    // ---- Prediction head ----
    gemm(h, KP2, predW1t, KP2, pred_b1, NH1, p1, KP1, 0, nullptr, 0, 0, NH1P / 16, 1);
    k_pred<<<(N + 7) / 8, 256, 0, stream>>>(p1, pred_W2, pred_b2, out, N);
}
